// Kernel_Layer_55654186221960
// MI455X (gfx1250) — matrix operand order repaired and checked by reading, not compiled, not run on hardware
//
#include <hip/hip_runtime.h>
#include <hip/hip_bf16.h>
#include <stdint.h>

// ---------------------------------------------------------------------------
// einsum('bmi,mio->bmo'):  16 independent GEMMs C_m = A_m(4096x512) * B_m(512x512)
// A/C row stride = 16*512 (x is (b,m,i)).  Memory-bound (~272MB @ 23.3TB/s),
// single pass: fp32 loads -> packed bf16 (v_perm_b32, 1 instr/pair) -> LDS
// (B transposed) -> v_wmma_f32_16x16x32_bf16, fp32 accumulate, fp32 epilogue.
// ---------------------------------------------------------------------------

#define M_DIM   16
#define D_IN    512
#define D_OUT   512
#define BATCH   4096

#define BT      128   // batch-rows per block tile
#define NT      128   // d_out cols per block tile
#define KT      32    // k per LDS tile (one WMMA K step)
#define LDS_STR 40    // ushort row stride (80B = multiple of 16B for b128 loads)
#define THREADS 256   // 8 waves

typedef __attribute__((ext_vector_type(16))) __bf16 v16bf;
typedef __attribute__((ext_vector_type(8)))  float  v8f;

union FragBF { v16bf v; uint4 q[2]; };

#if __has_builtin(__builtin_amdgcn_cvt_pk_bf16_f32)
typedef __attribute__((ext_vector_type(2))) __bf16 v2bf;
__device__ __forceinline__ uint32_t pk2bf(float lo, float hi) {
  v2bf r = __builtin_amdgcn_cvt_pk_bf16_f32(lo, hi);   // native packed cvt (RNE)
  return __builtin_bit_cast(uint32_t, r);
}
#else
__device__ __forceinline__ uint32_t pk2bf(float lo, float hi) {
  // round-half-up bf16, then v_perm_b32 packs both high halves in 1 instr:
  // dst = {hi.b3, hi.b2, lo.b3, lo.b2}  (sel bytes 4-7 -> S0=hi, 0-3 -> S1=lo)
  const uint32_t ul = __builtin_bit_cast(uint32_t, lo) + 0x8000u;
  const uint32_t uh = __builtin_bit_cast(uint32_t, hi) + 0x8000u;
  return __builtin_amdgcn_perm(uh, ul, 0x07060302u);
}
#endif

__global__ __launch_bounds__(THREADS, 1)
void Kernel_Layer_55654186221960_kernel(const float* __restrict__ x,
                                        const float* __restrict__ w,
                                        float* __restrict__ out)
{
  // Double-buffered tiles; B stored transposed ([n][k]) so B fragments are
  // contiguous-in-K b128 loads at compute time.
  __shared__ __align__(16) uint16_t As[2][BT][LDS_STR];  // [row][k]
  __shared__ __align__(16) uint16_t Bs[2][NT][LDS_STR];  // [n][k]

  const int tid   = threadIdx.x;
  const int lane  = tid & 31;
  const int wave  = tid >> 5;      // 0..7
  const int wm    = wave & 3;      // row group: rows [wm*32, wm*32+32)
  const int wn    = wave >> 2;     // col group: cols [wn*64, wn*64+64)
  const int mlane = lane & 15;
  const int lhalf = lane >> 4;     // lane half selects K sub-range (contiguous: k = 16*lhalf + s)

  const int b0 = blockIdx.x * BT;
  const int n0 = blockIdx.y * NT;
  const int m  = blockIdx.z;

  const float* Abase = x + ((size_t)b0 * M_DIM + m) * D_IN;      // row stride M_DIM*D_IN
  const float* Bbase = w + (size_t)m * D_IN * D_OUT + n0;        // row stride D_OUT

  // ---- per-thread global pointers, advanced by constant strides in k-loop --
  const float* aPtr[4];
  const float* bPtr[4];
  int aRow[4], aC4[4], bK[4], bC4[4];
  #pragma unroll
  for (int i = 0; i < 4; ++i) {
    const int lin = i * THREADS + tid;                   // 0..1023
    aRow[i] = lin >> 3;                                  // 0..127 (A: 128 rows x 8 float4)
    aC4 [i] = lin & 7;
    bK  [i] = lin >> 5;                                  // 0..31  (B: 32 k-rows x 32 float4)
    bC4 [i] = lin & 31;
    aPtr[i] = Abase + (size_t)aRow[i] * (M_DIM * D_IN) + aC4[i] * 4;
    bPtr[i] = Bbase + (size_t)bK[i] * D_OUT + bC4[i] * 4;
  }

  const v8f vzero = {0.f, 0.f, 0.f, 0.f, 0.f, 0.f, 0.f, 0.f};
  v8f acc[2][4];
  #pragma unroll
  for (int i = 0; i < 2; ++i)
    #pragma unroll
    for (int j = 0; j < 4; ++j)
      acc[i][j] = vzero;

  uint2 aReg[4], bReg[4];   // packed-bf16 staging for the software pipeline

  auto load_tiles = [&]() {
    #pragma unroll
    for (int i = 0; i < 4; ++i) {
      const float4 a = *(const float4*)aPtr[i];
      aReg[i].x = pk2bf(a.x, a.y);
      aReg[i].y = pk2bf(a.z, a.w);
      aPtr[i] += KT;                                     // next k-block, A
      const float4 b = *(const float4*)bPtr[i];
      bReg[i].x = pk2bf(b.x, b.y);
      bReg[i].y = pk2bf(b.z, b.w);
      bPtr[i] += (size_t)KT * D_OUT;                     // next k-block, B
    }
  };

  auto store_tiles = [&](int buf) {
    #pragma unroll
    for (int i = 0; i < 4; ++i) {
      *(uint2*)&As[buf][aRow[i]][aC4[i] * 4] = aReg[i];  // 8B-aligned
      Bs[buf][bC4[i] * 4 + 0][bK[i]] = (uint16_t)(bReg[i].x & 0xFFFFu);  // transpose
      Bs[buf][bC4[i] * 4 + 1][bK[i]] = (uint16_t)(bReg[i].x >> 16);
      Bs[buf][bC4[i] * 4 + 2][bK[i]] = (uint16_t)(bReg[i].y & 0xFFFFu);
      Bs[buf][bC4[i] * 4 + 3][bK[i]] = (uint16_t)(bReg[i].y >> 16);
    }
  };

  auto compute = [&](int buf) {
    FragBF afrag[2], bfrag[4];
    // 16-bit B 32x16 (order used here): lanes 0-15 hold K=0..15, lanes 16-31 K=16..31 (contiguous)
    #pragma unroll
    for (int j = 0; j < 4; ++j) {
      const int n  = wn * 64 + j * 16 + mlane;
      const int kb = lhalf * 16;
      bfrag[j].q[0] = *(const uint4*)&Bs[buf][n][kb];
      bfrag[j].q[1] = *(const uint4*)&Bs[buf][n][kb + 8];
    }
    // 16-bit A 16x32: lanes 0-15 K=0..15, lanes 16-31 K=16..31 (contiguous, matches B)
    #pragma unroll
    for (int i = 0; i < 2; ++i) {
      const int r  = wm * 32 + i * 16 + mlane;
      const int kb = lhalf * 16;
      afrag[i].q[0] = *(const uint4*)&As[buf][r][kb];
      afrag[i].q[1] = *(const uint4*)&As[buf][r][kb + 8];
    }
    #pragma unroll
    for (int i = 0; i < 2; ++i)
      #pragma unroll
      for (int j = 0; j < 4; ++j)
        acc[i][j] = __builtin_amdgcn_wmma_f32_16x16x32_bf16(
            false, afrag[i].v, false, bfrag[j].v,
            (short)0, acc[i][j], false, false);
  };

  // -------- software pipeline: prefetch k+1 into regs while computing k ----
  load_tiles();
  store_tiles(0);
  __syncthreads();

  const int KB = D_IN / KT;  // 16
  for (int kb = 0; kb < KB; ++kb) {
    if (kb + 1 < KB) load_tiles();
    compute(kb & 1);
    if (kb + 1 < KB) store_tiles((kb + 1) & 1);
    __syncthreads();
  }

  // -------- epilogue: fp32 store, ISA C/D layout (VGPR r -> row r / r+8) ---
  float* Cbase = out + ((size_t)b0 * M_DIM + m) * D_OUT + n0;
  #pragma unroll
  for (int i = 0; i < 2; ++i)
    #pragma unroll
    for (int j = 0; j < 4; ++j)
      #pragma unroll
      for (int r = 0; r < 8; ++r) {
        const int row = wm * 32 + i * 16 + r + lhalf * 8;
        const int col = wn * 64 + j * 16 + mlane;
        Cbase[(size_t)row * (M_DIM * D_OUT) + col] = acc[i][j][r];
      }
}

extern "C" void kernel_launch(void* const* d_in, const int* in_sizes, int n_in,
                              void* d_out, int out_size, void* d_ws, size_t ws_size,
                              hipStream_t stream) {
  const float* x = (const float*)d_in[0];   // (4096, 16, 512) fp32
  const float* w = (const float*)d_in[1];   // (16, 512, 512) fp32
  float* out = (float*)d_out;               // (4096, 16, 512) fp32

  dim3 grid(BATCH / BT, D_OUT / NT, M_DIM); // (32, 4, 16)
  dim3 block(THREADS);
  Kernel_Layer_55654186221960_kernel<<<grid, block, 0, stream>>>(x, w, out);
}
